// SelfAttention_21603685499269
// MI455X (gfx1250) — matrix operand order repaired and checked by reading, not compiled, not run on hardware
//
#include <hip/hip_runtime.h>

// ---------------------------------------------------------------------------
// Problem constants: B=4, N=2048, DIM=1024, H=16, HD=64, SCALE = 0.125
// ---------------------------------------------------------------------------
#define BATCH 4
#define SEQ   2048
#define DIM   1024
#define NH    16
#define HD    64
#define SCALE 0.125f

typedef __attribute__((ext_vector_type(8)))  float          v8f;
typedef __attribute__((ext_vector_type(16))) __bf16         v16bf;
typedef __attribute__((ext_vector_type(8)))  unsigned short u16x8;
typedef __attribute__((ext_vector_type(16))) unsigned short u16x16;
typedef __attribute__((ext_vector_type(4)))  int            i32x4;

// address-space-qualified int4 for the async-to-LDS builtin signature
typedef __attribute__((address_space(1))) i32x4 g_i32x4;  // global
typedef __attribute__((address_space(3))) i32x4 l_i32x4;  // LDS

// native f32 -> bf16 conversion (round-to-nearest-even via hardware cvt)
static __device__ __forceinline__ unsigned short f2bfu(float f) {
    return __builtin_bit_cast(unsigned short, (__bf16)f);
}

static __device__ __forceinline__ v16bf make_frag(u16x8 lo, u16x8 hi) {
    union { u16x8 h[2]; u16x16 w; } u;
    u.h[0] = lo; u.h[1] = hi;
    return __builtin_bit_cast(v16bf, u.w);
}

static __device__ __forceinline__ v16bf frag32(const unsigned short* p) {
    return __builtin_bit_cast(v16bf, *(const u16x16*)p);
}

static __device__ __forceinline__ v8f wmma_bf16(v16bf a, v16bf b, v8f c) {
    return __builtin_amdgcn_wmma_f32_16x16x32_bf16(
        false, a, false, b, (short)0, c, false, false);
}

// ---------------------------------------------------------------------------
// CDNA5 async global->LDS copy (16B per lane, per-lane addresses), tracked by
// ASYNCcnt. Guarded by __has_builtin with a synchronous fallback.
// ---------------------------------------------------------------------------
#if defined(__has_builtin)
#  if __has_builtin(__builtin_amdgcn_global_load_async_to_lds_b128)
#    define HAVE_ASYNC_LDS 1
#  endif
#endif
#ifndef HAVE_ASYNC_LDS
#  define HAVE_ASYNC_LDS 0
#endif

static __device__ __forceinline__ void async_cp16(const unsigned short* g,
                                                  unsigned short* l) {
#if HAVE_ASYNC_LDS
    __builtin_amdgcn_global_load_async_to_lds_b128(
        (g_i32x4*)g, (l_i32x4*)l, 0, 0);
#else
    *(u16x8*)l = *(const u16x8*)g;
#endif
}

static __device__ __forceinline__ void async_wait() {
#if HAVE_ASYNC_LDS
#  if defined(__has_builtin) && __has_builtin(__builtin_amdgcn_s_wait_asynccnt)
    __builtin_amdgcn_s_wait_asynccnt(0);
#  else
    asm volatile("s_wait_asynccnt 0x0" ::: "memory");
#  endif
#endif
}

// ---------------------------------------------------------------------------
// Kernel 0: f32 -> bf16 convert (optionally transposing [rows][cols] ->
// [cols][rows]) so every later tile stage is a linear bf16 async copy.
// ---------------------------------------------------------------------------
__global__ __launch_bounds__(256) void cvt_kernel(
    const float* __restrict__ src, unsigned short* __restrict__ dst,
    int rows, int cols, int transpose)
{
    const int i = blockIdx.x * 256 + threadIdx.x;
    if (i >= rows * cols) return;
    const int r = i / cols, c = i - r * cols;
    const unsigned short v = f2bfu(src[i]);
    if (transpose) dst[(size_t)c * rows + r] = v;
    else           dst[i] = v;
}

// ---------------------------------------------------------------------------
// Kernel 1: QKV GEMM (M=8192,K=1024,N=3072), bf16 A (x_bf) x bf16 B (WqkvT),
// async LDS staging, WMMA bf16, epilogue scatters Q(scaled)/K row-major and
// V head-transposed [b][h][hd][n].
// ---------------------------------------------------------------------------
__global__ __launch_bounds__(256) void qkv_gemm_kernel(
    const unsigned short* __restrict__ xbf,
    const unsigned short* __restrict__ WqkvT,
    const float* __restrict__ bqkv,
    unsigned short* __restrict__ qbf, unsigned short* __restrict__ kbf,
    unsigned short* __restrict__ vbf)
{
    __shared__ __align__(32) unsigned short As[128][32]; // [row][k]
    __shared__ __align__(32) unsigned short Bs[128][32]; // [col][k]

    const int tid   = threadIdx.x;
    const int lane  = tid & 31;
    const int wid   = tid >> 5;
    const int khalf = lane >> 4;
    const int ln    = lane & 15;
    const int r0    = blockIdx.y * 128;
    const int c0    = blockIdx.x * 128;
    const int wrow  = wid >> 1;
    const int wcol  = wid & 1;

    v8f acc[2][4];
    #pragma unroll
    for (int i = 0; i < 2; i++)
        #pragma unroll
        for (int j = 0; j < 4; j++)
            acc[i][j] = (v8f){};

    const int ldRow = tid >> 1;        // 0..127 (row for A, col for B)
    const int ldOff = (tid & 1) * 16;  // 0 / 16

    for (int kk = 0; kk < DIM; kk += 32) {
        // A tile: xbf[r0+row][kk .. kk+31]  (linear bf16 -> async copy)
        {
            const unsigned short* gp = xbf + (size_t)(r0 + ldRow) * DIM + kk + ldOff;
            async_cp16(gp,     &As[ldRow][ldOff]);
            async_cp16(gp + 8, &As[ldRow][ldOff + 8]);
        }
        // B tile: WqkvT[c0+col][kk .. kk+31]  (pre-transposed -> linear)
        {
            const unsigned short* gp = WqkvT + (size_t)(c0 + ldRow) * DIM + kk + ldOff;
            async_cp16(gp,     &Bs[ldRow][ldOff]);
            async_cp16(gp + 8, &Bs[ldRow][ldOff + 8]);
        }
        async_wait();
        __syncthreads();

        v16bf afr[2];
        #pragma unroll
        for (int rt = 0; rt < 2; rt++) {
            const unsigned short* pa = &As[wrow * 32 + rt * 16 + ln][0];
            afr[rt] = make_frag(*(const u16x8*)(pa + khalf * 16),
                                *(const u16x8*)(pa + khalf * 16 + 8));
        }
        #pragma unroll
        for (int ct = 0; ct < 4; ct++) {
            v16bf bfr = frag32(&Bs[wcol * 64 + ct * 16 + ln][khalf * 16]);
            acc[0][ct] = wmma_bf16(afr[0], bfr, acc[0][ct]);
            acc[1][ct] = wmma_bf16(afr[1], bfr, acc[1][ct]);
        }
        __syncthreads();
    }

    #pragma unroll
    for (int rt = 0; rt < 2; rt++) {
        #pragma unroll
        for (int ct = 0; ct < 4; ct++) {
            const int colg  = c0 + wcol * 64 + ct * 16 + ln;
            const int which = colg >> 10;          // 0=Q,1=K,2=V
            const int d     = colg & (DIM - 1);
            const int hh    = d >> 6;
            const int hd    = d & (HD - 1);
            const float bias = bqkv[colg];
            const float scl  = (which == 0) ? SCALE : 1.0f;
            #pragma unroll
            for (int r = 0; r < 8; r++) {
                const int rowg = r0 + wrow * 32 + rt * 16 + r + 8 * khalf;
                const int bb   = rowg >> 11;
                const int n    = rowg & (SEQ - 1);
                const unsigned short v = f2bfu((acc[rt][ct][r] + bias) * scl);
                const size_t head = (size_t)(bb * NH + hh);
                if (which == 0)      qbf[(head * SEQ + n) * HD + hd] = v;
                else if (which == 1) kbf[(head * SEQ + n) * HD + hd] = v;
                else                 vbf[(head * HD + hd) * SEQ + n] = v; // transposed
            }
        }
    }
}

// ---------------------------------------------------------------------------
// Kernel 2: flash attention. KV tiles staged via async copies (K row-major,
// V already transposed in workspace -> linear).
// ---------------------------------------------------------------------------
__global__ __launch_bounds__(256) void attn_kernel(
    const unsigned short* __restrict__ qbf,
    const unsigned short* __restrict__ kbf,
    const unsigned short* __restrict__ vbf,
    unsigned short* __restrict__ attn)
{
    __shared__ __align__(32) unsigned short Ks[64][HD];     // [key][dim]
    __shared__ __align__(32) unsigned short Vt[HD][64];     // [dim][key]
    __shared__ __align__(32) unsigned short Ps[8][16][64];  // per-wave P tile

    const int tid   = threadIdx.x;
    const int lane  = tid & 31;
    const int wid   = tid >> 5;
    const int khalf = lane >> 4;
    const int ln    = lane & 15;
    const int qt    = blockIdx.x;
    const int bh    = blockIdx.y;            // b*NH + h
    const int h     = bh & (NH - 1);
    const int b     = bh >> 4;

    const size_t headRM = (size_t)bh * SEQ * HD;  // Q/K row-major head base
    const size_t headTR = (size_t)bh * HD * SEQ;  // V transposed head base

    const int n0 = qt * 128 + wid * 16;

    // preload Q A-fragments (already scaled)
    v16bf qa[2];
    {
        const unsigned short* qp = qbf + headRM + (size_t)(n0 + ln) * HD;
        #pragma unroll
        for (int kc = 0; kc < 2; kc++) {
            u16x8 lo = *(const u16x8*)(qp + kc * 32 + khalf * 16);
            u16x8 hi = *(const u16x8*)(qp + kc * 32 + khalf * 16 + 8);
            qa[kc] = make_frag(lo, hi);
        }
    }

    v8f o[4];
    #pragma unroll
    for (int oc = 0; oc < 4; oc++) o[oc] = (v8f){};
    float mprev[8], lsum[8];
    #pragma unroll
    for (int r = 0; r < 8; r++) { mprev[r] = -1e30f; lsum[r] = 0.0f; }

    const int lrow = tid >> 2;          // 0..63 (key for K, dim for V)
    const int lcb  = (tid & 3) * 16;    // 0,16,32,48

    for (int kv0 = 0; kv0 < SEQ; kv0 += 64) {
        __syncthreads();
        // K tile: Ks[key][.] <- kbf[head][kv0+key][.]  (linear, async)
        {
            const unsigned short* gp = kbf + headRM + (size_t)(kv0 + lrow) * HD + lcb;
            async_cp16(gp,     &Ks[lrow][lcb]);
            async_cp16(gp + 8, &Ks[lrow][lcb + 8]);
        }
        // V tile: Vt[dim][.] <- vbf[head][dim][kv0 + .]  (linear, async)
        {
            const unsigned short* gp = vbf + headTR + (size_t)lrow * SEQ + kv0 + lcb;
            async_cp16(gp,     &Vt[lrow][lcb]);
            async_cp16(gp + 8, &Vt[lrow][lcb + 8]);
        }
        async_wait();
        __syncthreads();

        // S = Q @ K^T
        v8f s[4];
        #pragma unroll
        for (int ct = 0; ct < 4; ct++) {
            v8f z = (v8f){};
            v16bf b0 = frag32(&Ks[ct * 16 + ln][khalf * 16]);
            v16bf b1 = frag32(&Ks[ct * 16 + ln][32 + khalf * 16]);
            z     = wmma_bf16(qa[0], b0, z);
            s[ct] = wmma_bf16(qa[1], b1, z);
        }

        // online softmax over this 64-key tile
        #pragma unroll
        for (int r = 0; r < 8; r++) {
            float mn = fmaxf(fmaxf(s[0][r], s[1][r]), fmaxf(s[2][r], s[3][r]));
            mn = fmaxf(mn, __shfl_xor(mn, 1, 32));
            mn = fmaxf(mn, __shfl_xor(mn, 2, 32));
            mn = fmaxf(mn, __shfl_xor(mn, 4, 32));
            mn = fmaxf(mn, __shfl_xor(mn, 8, 32));
            const float mi    = fmaxf(mprev[r], mn);
            const float alpha = __expf(mprev[r] - mi);
            mprev[r] = mi;

            float rs = 0.0f;
            const int prow = r + 8 * khalf;
            #pragma unroll
            for (int ct = 0; ct < 4; ct++) {
                const float p = __expf(s[ct][r] - mi);
                rs += p;
                Ps[wid][prow][ct * 16 + ln] = f2bfu(p);
            }
            rs += __shfl_xor(rs, 1, 32);
            rs += __shfl_xor(rs, 2, 32);
            rs += __shfl_xor(rs, 4, 32);
            rs += __shfl_xor(rs, 8, 32);
            lsum[r] = lsum[r] * alpha + rs;
            #pragma unroll
            for (int oc = 0; oc < 4; oc++) o[oc][r] *= alpha;
        }

        // O += P @ V
        #pragma unroll
        for (int kc = 0; kc < 2; kc++) {
            const unsigned short* pp = &Ps[wid][ln][kc * 32];
            v16bf pa = make_frag(*(const u16x8*)(pp + khalf * 16),
                                 *(const u16x8*)(pp + khalf * 16 + 8));
            #pragma unroll
            for (int oc = 0; oc < 4; oc++) {
                v16bf vb = frag32(&Vt[oc * 16 + ln][kc * 32 + khalf * 16]);
                o[oc] = wmma_bf16(pa, vb, o[oc]);
            }
        }
    }

    // epilogue: normalize + faithful-reshape store
    #pragma unroll
    for (int r = 0; r < 8; r++) {
        const float inv = 1.0f / lsum[r];
        const int n   = n0 + r + 8 * khalf;
        const int m   = h * SEQ + n;
        const int n2  = m >> 4;
        const int d2b = (m & 15) * HD;
        #pragma unroll
        for (int oc = 0; oc < 4; oc++) {
            const int d2 = d2b + oc * 16 + ln;
            attn[((size_t)b * SEQ + n2) * DIM + d2] = f2bfu(o[oc][r] * inv);
        }
    }
}

// ---------------------------------------------------------------------------
// Kernel 3: out = attn @ W_proj + b_proj (M=8192,K=1024,N=1024), f32 out.
// Both operands bf16 in workspace (W pre-transposed) -> pure async staging.
// ---------------------------------------------------------------------------
__global__ __launch_bounds__(256) void proj_gemm_kernel(
    const unsigned short* __restrict__ attn,
    const unsigned short* __restrict__ WpT,
    const float* __restrict__ bp, float* __restrict__ out)
{
    __shared__ __align__(32) unsigned short As[128][32];
    __shared__ __align__(32) unsigned short Bs[128][32];

    const int tid   = threadIdx.x;
    const int lane  = tid & 31;
    const int wid   = tid >> 5;
    const int khalf = lane >> 4;
    const int ln    = lane & 15;
    const int r0    = blockIdx.y * 128;
    const int c0    = blockIdx.x * 128;
    const int wrow  = wid >> 1;
    const int wcol  = wid & 1;

    v8f acc[2][4];
    #pragma unroll
    for (int i = 0; i < 2; i++)
        #pragma unroll
        for (int j = 0; j < 4; j++)
            acc[i][j] = (v8f){};

    const int ldRow = tid >> 1;
    const int ldOff = (tid & 1) * 16;

    for (int kk = 0; kk < DIM; kk += 32) {
        {
            const unsigned short* gp = attn + (size_t)(r0 + ldRow) * DIM + kk + ldOff;
            async_cp16(gp,     &As[ldRow][ldOff]);
            async_cp16(gp + 8, &As[ldRow][ldOff + 8]);
        }
        {
            const unsigned short* gp = WpT + (size_t)(c0 + ldRow) * DIM + kk + ldOff;
            async_cp16(gp,     &Bs[ldRow][ldOff]);
            async_cp16(gp + 8, &Bs[ldRow][ldOff + 8]);
        }
        async_wait();
        __syncthreads();

        v16bf afr[2];
        #pragma unroll
        for (int rt = 0; rt < 2; rt++) {
            const unsigned short* pa = &As[wrow * 32 + rt * 16 + ln][0];
            afr[rt] = make_frag(*(const u16x8*)(pa + khalf * 16),
                                *(const u16x8*)(pa + khalf * 16 + 8));
        }
        #pragma unroll
        for (int ct = 0; ct < 4; ct++) {
            v16bf bfr = frag32(&Bs[wcol * 64 + ct * 16 + ln][khalf * 16]);
            acc[0][ct] = wmma_bf16(afr[0], bfr, acc[0][ct]);
            acc[1][ct] = wmma_bf16(afr[1], bfr, acc[1][ct]);
        }
        __syncthreads();
    }

    #pragma unroll
    for (int rt = 0; rt < 2; rt++) {
        #pragma unroll
        for (int ct = 0; ct < 4; ct++) {
            const int colg = c0 + wcol * 64 + ct * 16 + ln;
            const float bias = bp[colg];
            #pragma unroll
            for (int r = 0; r < 8; r++) {
                const int rowg = r0 + wrow * 32 + rt * 16 + r + 8 * khalf;
                out[(size_t)rowg * DIM + colg] = acc[rt][ct][r] + bias;
            }
        }
    }
}

// ---------------------------------------------------------------------------
// Workspace layout (bf16 elements):
//   qbf  : B*H*N*HD      = 8388608
//   kbf  : 8388608
//   vbf  : 8388608  (head-transposed [b][h][hd][n])
//   attn : B*N*DIM       = 8388608
//   xbf  : 8388608
//   WqkvT: 3*DIM*DIM     = 3145728 ([col][k])
//   WpT  : DIM*DIM       = 1048576 ([col][k])
// Total = 46,137,344 elems * 2B ~= 88 MB
// ---------------------------------------------------------------------------
extern "C" void kernel_launch(void* const* d_in, const int* in_sizes, int n_in,
                              void* d_out, int out_size, void* d_ws, size_t ws_size,
                              hipStream_t stream) {
    (void)in_sizes; (void)n_in; (void)out_size; (void)ws_size;
    const float* x    = (const float*)d_in[0];
    const float* Wqkv = (const float*)d_in[1];
    const float* bqkv = (const float*)d_in[2];
    const float* Wp   = (const float*)d_in[3];
    const float* bp   = (const float*)d_in[4];
    float* out = (float*)d_out;

    const size_t HEADSZ = (size_t)BATCH * NH * SEQ * HD;   // 8388608
    unsigned short* qbf   = (unsigned short*)d_ws;
    unsigned short* kbf   = qbf   + HEADSZ;
    unsigned short* vbf   = kbf   + HEADSZ;
    unsigned short* attn  = vbf   + HEADSZ;
    unsigned short* xbf   = attn  + HEADSZ;
    unsigned short* WqkvT = xbf   + HEADSZ;
    unsigned short* WpT   = WqkvT + (size_t)3 * DIM * DIM;

    dim3 blk(256, 1, 1);

    // stage bf16 copies (W matrices pre-transposed to [col][k])
    cvt_kernel<<<dim3((BATCH * SEQ * DIM) / 256), blk, 0, stream>>>(
        x, xbf, BATCH * SEQ, DIM, 0);
    cvt_kernel<<<dim3((DIM * 3 * DIM) / 256), blk, 0, stream>>>(
        Wqkv, WqkvT, DIM, 3 * DIM, 1);
    cvt_kernel<<<dim3((DIM * DIM) / 256), blk, 0, stream>>>(
        Wp, WpT, DIM, DIM, 1);

    qkv_gemm_kernel<<<dim3(3 * DIM / 128, BATCH * SEQ / 128), blk, 0, stream>>>(
        xbf, WqkvT, bqkv, qbf, kbf, vbf);
    attn_kernel<<<dim3(SEQ / 128, BATCH * NH), blk, 0, stream>>>(
        qbf, kbf, vbf, attn);
    proj_gemm_kernel<<<dim3(DIM / 128, BATCH * SEQ / 128), blk, 0, stream>>>(
        attn, WpT, bp, out);
}
